// Att_65085934403930
// MI455X (gfx1250) — matrix operand order repaired and checked by reading, not compiled, not run on hardware
//
#include <hip/hip_runtime.h>
#include <hip/hip_bf16.h>

#define EPSV 1e-7f

namespace {
constexpr int C    = 256;
constexpr int HW   = 4096;        // h*w = 64*64
constexpr int PADW = 66;
constexpr int PP   = PADW * PADW; // 4356
constexpr int PT   = 64;          // pixels per workgroup (16 per wave, 4 waves)
constexpr int JT   = 32;          // j-tile (one K=32 WMMA chunk for GEMM2)
constexpr int LDF  = 272;         // fT row stride in halfs (16B-aligned rows, conflict-free)
constexpr int LDA  = 272;         // kA row stride
constexpr int LDB  = 40;          // kB row stride (32 + 8 pad, 16B-aligned rows)
constexpr int LDP  = 40;          // attn tile row stride
constexpr int SM_HALFS = JT * LDA + C * LDB + 4 * 16 * LDP; // 21504 halfs = 43KB
}

typedef __attribute__((ext_vector_type(16))) _Float16 v16h;
typedef __attribute__((ext_vector_type(8)))  _Float16 v8h;
typedef __attribute__((ext_vector_type(4)))  _Float16 v4h;
typedef __attribute__((ext_vector_type(8)))  float    v8f;

union V16U { v16h v; v8h h[2]; };

// rnorm[b][j] = rsqrt(sum_c (inputs[b][c][j] + eps)^2)
__global__ __launch_bounds__(128) void rnorm_kernel(const float* __restrict__ in,
                                                    float* __restrict__ rn) {
  int g = blockIdx.x * 128 + threadIdx.x;  // 0 .. 4*4096-1
  int b = g >> 12;
  int j = g & (HW - 1);
  const float* p = in + (size_t)b * C * HW + j;
  float s = 0.f;
#pragma unroll 8
  for (int c = 0; c < C; ++c) {
    float v = p[(size_t)c * HW] + EPSV;
    s += v * v;
  }
  rn[g] = rsqrtf(s);
}

// Fused attention: S = k_norm @ f + gate(interior); softmax over j; out = k^T @ attn
__global__ __launch_bounds__(128) void attn_kernel(const float* __restrict__ in,
                                                   const float* __restrict__ gate,
                                                   const float* __restrict__ rn,
                                                   float* __restrict__ out) {
  __shared__ __align__(16) _Float16 sm[SM_HALFS];
  _Float16* fT = sm;                        // [PT][LDF] staging (dead after frag extract)
  _Float16* kA = sm;                        // [JT][LDA]  k * rnorm   (GEMM1 A)
  _Float16* kB = sm + JT * LDA;             // [C][LDB]   raw k       (GEMM2 A)
  _Float16* aT = sm + JT * LDA + C * LDB;   // [4][16][LDP] attn tile (GEMM2 B^T), per wave

  const int tid  = threadIdx.x;
  const int wid  = tid >> 5;
  const int lane = tid & 31;
  const int l    = lane & 15;
  const int hig  = lane >> 4;   // 0 / 1
  const int hi8  = hig << 3;
  const int hi16 = hig << 4;

  const int b  = blockIdx.y;
  const int p0 = blockIdx.x * PT;

  const float* inB = in + (size_t)b * C * HW;

  // ---- stage f-tile (256c x 64p) into LDS as [p][c] fp16 ----
  for (int it = 0; it < 32; ++it) {
    int idx = it * 128 + tid;           // 0 .. PT*C/4-1
    int c  = idx >> 4;
    int pq = (idx & 15) << 2;
    float4 v = *(const float4*)(inB + (size_t)c * HW + p0 + pq);
    fT[(pq + 0) * LDF + c] = (_Float16)v.x;
    fT[(pq + 1) * LDF + c] = (_Float16)v.y;
    fT[(pq + 2) * LDF + c] = (_Float16)v.z;
    fT[(pq + 3) * LDF + c] = (_Float16)v.w;
  }
  __syncthreads();

  // resident GEMM1 B fragments: f as (K=c 32-chunk) x (N=p 16), loaded from B^T=[p][c]
  v16h fB[8];
#pragma unroll
  for (int cc = 0; cc < 8; ++cc) {
    const _Float16* bp = fT + (wid * 16 + l) * LDF + cc * 32 + hi8;
    V16U u;
    u.h[0] = *(const v8h*)(bp);
    u.h[1] = *(const v8h*)(bp + 16);
    fB[cc] = u.v;
  }

  v8f acc[16];
#pragma unroll
  for (int i = 0; i < 16; ++i)
#pragma unroll
    for (int r = 0; r < 8; ++r) acc[i][r] = 0.f;

  float m = -1e30f, ssum = 0.f;

  const int pcol = p0 + wid * 16 + l;  // this lane's pixel column
  const float* gcol = gate + (size_t)b * HW * PP
                    + (size_t)(pcol >> 6) * PADW + (pcol & 63) + PADW + 1;
  const float* rnB = rn + (b << 12);

  for (int j0 = 0; j0 < HW; j0 += JT) {
    __syncthreads();  // previous tile's kA/kB reads (or fT) complete

    // cooperative k-tile load: kB = k (raw, [c][j]), kA = k*rnorm ([j][c])
    for (int it = 0; it < 16; ++it) {
      int idx = it * 128 + tid;        // 0 .. C*JT/4-1
      int c  = idx >> 3;
      int jq = (idx & 7) << 2;
      float4 v = *(const float4*)(inB + (size_t)c * HW + j0 + jq);
      float x0 = v.x + EPSV, x1 = v.y + EPSV, x2 = v.z + EPSV, x3 = v.w + EPSV;
      v4h kv = { (_Float16)x0, (_Float16)x1, (_Float16)x2, (_Float16)x3 };
      *(v4h*)(kB + c * LDB + jq) = kv;
      kA[(jq + 0) * LDA + c] = (_Float16)(x0 * rnB[j0 + jq + 0]);
      kA[(jq + 1) * LDA + c] = (_Float16)(x1 * rnB[j0 + jq + 1]);
      kA[(jq + 2) * LDA + c] = (_Float16)(x2 * rnB[j0 + jq + 2]);
      kA[(jq + 3) * LDA + c] = (_Float16)(x3 * rnB[j0 + jq + 3]);
    }
    __syncthreads();

    // prefetch next j-tile's k rows into cache while we compute (1 line = 32 j each)
    {
      int jn = j0 + JT;
      if (jn < HW) {
        __builtin_prefetch(inB + (size_t)tid * HW + jn, 0, 1);
        __builtin_prefetch(inB + (size_t)(tid + 128) * HW + jn, 0, 1);
      }
    }

    // ---- GEMM1: S(j,p) = k_norm(j,:) . f(:,p), f32 accumulate ----
    v8f S[2];
#pragma unroll
    for (int jj = 0; jj < 2; ++jj)
#pragma unroll
      for (int r = 0; r < 8; ++r) S[jj][r] = 0.f;

    auto loadA1 = [&](int i) -> V16U {   // i = cc*2 + jj
      int jj = i & 1, cc = i >> 1;
      const _Float16* ap = kA + (jj * 16 + l) * LDA + cc * 32 + hi8;
      V16U u;
      u.h[0] = *(const v8h*)(ap);
      u.h[1] = *(const v8h*)(ap + 16);
      return u;
    };
    // 2-deep pipelined fragment ring: frag i+1, i+2 in flight while WMMA consumes i
    V16U af[2];
    af[0] = loadA1(0);
    af[1] = loadA1(1);
#pragma unroll
    for (int i = 0; i < 16; ++i) {
      v16h cur = af[i & 1].v;
      if (i + 2 < 16) af[i & 1] = loadA1(i + 2);
      S[i & 1] = __builtin_amdgcn_wmma_f32_16x16x32_f16(
          false, cur, false, fB[i >> 1], (short)0, S[i & 1], false, false);
    }

    // + interior gate scores (row j = j0 + jj*16 + hi8 + r, col = pcol)
#pragma unroll
    for (int jj = 0; jj < 2; ++jj)
#pragma unroll
      for (int r = 0; r < 8; ++r)
        S[jj][r] += gcol[(size_t)(j0 + jj * 16 + hi8 + r) * PP];

    // ---- online softmax over j for column p = lane%16 ----
    float mt = -1e30f;
#pragma unroll
    for (int jj = 0; jj < 2; ++jj)
#pragma unroll
      for (int r = 0; r < 8; ++r) mt = fmaxf(mt, S[jj][r]);
    mt = fmaxf(mt, __shfl_xor(mt, 16));
    float mnew  = fmaxf(m, mt);
    float alpha = __expf(m - mnew);
    float pt = 0.f;
#pragma unroll
    for (int jj = 0; jj < 2; ++jj)
#pragma unroll
      for (int r = 0; r < 8; ++r) {
        float e = __expf(S[jj][r] - mnew);
        S[jj][r] = e;
        pt += e;
      }
    pt += __shfl_xor(pt, 16);
    ssum = ssum * alpha + pt;
    m = mnew;

#pragma unroll
    for (int i = 0; i < 16; ++i)
#pragma unroll
      for (int r = 0; r < 8; ++r) acc[i][r] *= alpha;

    // stash attn tile fp16 as [p][j] (per-wave region) for GEMM2 B operand
    _Float16* aw = aT + wid * (16 * LDP);
#pragma unroll
    for (int jj = 0; jj < 2; ++jj) {
      v8h hh;
#pragma unroll
      for (int r = 0; r < 8; ++r) hh[r] = (_Float16)S[jj][r];
      *(v8h*)(aw + l * LDP + jj * 16 + hi8) = hh;
    }

    // ---- GEMM2: acc(c,p) += k(j,c)^T @ attn(j,p), single K=32 chunk ----
    auto loadV2 = [&](int cf) -> V16U {
      const _Float16* vp = kB + (cf * 16 + l) * LDB + hi8;
      V16U u;
      u.h[0] = *(const v8h*)(vp);
      u.h[1] = *(const v8h*)(vp + 16);
      return u;
    };
    V16U pu;
    {
      const _Float16* pb = aw + l * LDP + hi8;
      pu.h[0] = *(const v8h*)(pb);
      pu.h[1] = *(const v8h*)(pb + 16);
    }
    V16U vf[2];
    vf[0] = loadV2(0);
    vf[1] = loadV2(1);
#pragma unroll
    for (int cf = 0; cf < 16; ++cf) {
      v16h cur = vf[cf & 1].v;
      if (cf + 2 < 16) vf[cf & 1] = loadV2(cf + 2);
      acc[cf] = __builtin_amdgcn_wmma_f32_16x16x32_f16(
          false, cur, false, pu.v, (short)0, acc[cf], false, false);
    }
  }

  // ---- epilogue: divide by softmax sum, store f32 ----
  float inv = 1.f / ssum;
  float* outB = out + (size_t)b * C * HW;
#pragma unroll
  for (int cf = 0; cf < 16; ++cf)
#pragma unroll
    for (int r = 0; r < 8; ++r)
      outB[(size_t)(cf * 16 + hi8 + r) * HW + pcol] = acc[cf][r] * inv;
}

extern "C" void kernel_launch(void* const* d_in, const int* in_sizes, int n_in,
                              void* d_out, int out_size, void* d_ws, size_t ws_size,
                              hipStream_t stream) {
  (void)in_sizes; (void)n_in; (void)out_size; (void)ws_size;
  const float* in   = (const float*)d_in[0];
  const float* gate = (const float*)d_in[1];
  float* rnormv = (float*)d_ws;   // 4*4096 floats = 64 KB scratch
  float* outp   = (float*)d_out;

  rnorm_kernel<<<dim3((4 * HW) / 128), 128, 0, stream>>>(in, rnormv);
  attn_kernel<<<dim3(HW / PT, 4), 128, 0, stream>>>(in, gate, rnormv, outp);
}
